// SelfAttention_89515708383624
// MI455X (gfx1250) — matrix operand order repaired and checked by reading, compiled, not run on hardware
//
#include <hip/hip_runtime.h>

// Problem constants (from reference setup)
#define BB 4
#define CC 256
#define C8 32
#define NN 4096   // 64*64

#define LROW 40           // LDS row stride in bf16 elements (80 B, 16B-aligned, bank-conflict-free)
#define NCHUNK (NN / 32)  // 128 j-chunks of 32

typedef __attribute__((ext_vector_type(16))) __bf16 v16bf;
typedef __attribute__((ext_vector_type(8)))  float  v8f;
typedef int gvec4i __attribute__((vector_size(16)));   // type expected by async builtin

union V16U { v16bf v; unsigned u[8]; uint4 q4[2]; };

__device__ __forceinline__ unsigned short f2bf(float f) {
  unsigned u = __float_as_uint(f);
  u += 0x7FFFu + ((u >> 16) & 1u);          // round-to-nearest-even
  return (unsigned short)(u >> 16);
}
__device__ __forceinline__ unsigned pack2bf(float lo, float hi) {
  return (unsigned)f2bf(lo) | ((unsigned)f2bf(hi) << 16);
}

// ---- CDNA5 async global->LDS copy (ASYNCcnt path), guarded fallbacks ------
__device__ __forceinline__ void async_cp16(const unsigned short* g, unsigned short* l) {
#if __has_builtin(__builtin_amdgcn_global_load_async_to_lds_b128)
  __builtin_amdgcn_global_load_async_to_lds_b128(
      (__attribute__((address_space(1))) gvec4i*)(unsigned short*)g,
      (__attribute__((address_space(3))) gvec4i*)l, 0, 0);
#else
  *(uint4*)l = *(const uint4*)g;   // sync fallback: global->VGPR->ds_store
#endif
}
__device__ __forceinline__ void async_wait0() {
#if __has_builtin(__builtin_amdgcn_global_load_async_to_lds_b128)
#if __has_builtin(__builtin_amdgcn_s_wait_asynccnt)
  __builtin_amdgcn_s_wait_asynccnt(0);
#else
  asm volatile("s_wait_asynccnt 0" ::: "memory");
#endif
#endif
}

// ---------------------------------------------------------------------------
// Stage 1: q/k/v 1x1-conv projections (K=256 GEMMs, ~7% of FLOPs) -> bf16.
//   q stored [B, N, 32]  (row n = q[n,:])   -> WMMA B operand
//   k stored [B, N, 32]  (row j = k[:,j])   -> WMMA A operand
//   v stored [B, C, N]   (row c = v[c,:])   -> WMMA A operand
// ---------------------------------------------------------------------------
__global__ void __launch_bounds__(256)
qkv_project(const float* __restrict__ x,
            const float* __restrict__ wq, const float* __restrict__ bq,
            const float* __restrict__ wk, const float* __restrict__ bk,
            const float* __restrict__ wv, const float* __restrict__ bv,
            unsigned short* __restrict__ qbf,
            unsigned short* __restrict__ kbf,
            unsigned short* __restrict__ vbf) {
  __shared__ float wsm[8 * CC];
  const int tid    = threadIdx.x;
  const int nchunk = blockIdx.x;   // 16
  const int og     = blockIdx.y;   // 40 groups of 8 outputs (32 q + 32 k + 256 v)
  const int b      = blockIdx.z;   // 4
  const int o0     = og * 8;

  const float* wsrc; const float* bsrc;
  if (o0 < 32)      { wsrc = wq + (size_t)o0 * CC;        bsrc = bq + o0; }
  else if (o0 < 64) { wsrc = wk + (size_t)(o0 - 32) * CC; bsrc = bk + (o0 - 32); }
  else              { wsrc = wv + (size_t)(o0 - 64) * CC; bsrc = bv + (o0 - 64); }

  for (int i = tid; i < 8 * CC; i += 256) wsm[i] = wsrc[i];
  __syncthreads();

  const int n = nchunk * 256 + tid;
  const float* xp = x + (size_t)b * CC * NN + n;   // x[b, c, n], c-stride = N

  float acc[8];
  #pragma unroll
  for (int i = 0; i < 8; ++i) acc[i] = bsrc[i];

  for (int c = 0; c < CC; ++c) {
    float xv = xp[(size_t)c * NN];                 // coalesced across tid
    #pragma unroll
    for (int i = 0; i < 8; ++i) acc[i] = fmaf(wsm[i * CC + c], xv, acc[i]);
  }

  if (o0 < 32) {
    #pragma unroll
    for (int i = 0; i < 8; ++i)
      qbf[((size_t)(b * NN + n)) * C8 + o0 + i] = f2bf(acc[i]);
  } else if (o0 < 64) {
    #pragma unroll
    for (int i = 0; i < 8; ++i)
      kbf[((size_t)(b * NN + n)) * C8 + (o0 - 32) + i] = f2bf(acc[i]);
  } else {
    #pragma unroll
    for (int i = 0; i < 8; ++i)
      vbf[((size_t)(b * CC + (o0 - 64) + i)) * NN + n] = f2bf(acc[i]);
  }
}

// ---------------------------------------------------------------------------
// Stage 2: fused flash attention + residual.
// Workgroup = 256 threads = 8 waves: wave w -> query tile (w&3), channel half (w>>2).
// Per 32-j chunk, the whole workgroup async-DMAs K(32x32) + V(256x32) bf16
// into double-buffered LDS (padded rows), overlapped with WMMA compute.
//  S^T tile:  A = K-tile (M=j, K=ch32), B = q^T  -> queries in lanes, j in VGPRs
//  O tile:    A = V-tile (M=c, K=j32),  B = P^T  -> accumulated in v8f regs
// ---------------------------------------------------------------------------
__global__ void __launch_bounds__(256)
flash_attn(const unsigned short* __restrict__ qbf,
           const unsigned short* __restrict__ kbf,
           const unsigned short* __restrict__ vbf,
           const float* __restrict__ x,
           const float* __restrict__ gamma,
           float* __restrict__ out) {
  __shared__ __align__(16) unsigned short Kb[2][32 * LROW];    //  5 KB
  __shared__ __align__(16) unsigned short Vb[2][CC * LROW];    // 40 KB

  const int tid  = threadIdx.x;
  const int lane = tid & 31;
  const int wv_  = tid >> 5;        // wave id 0..7
  const int hl   = lane >> 4;       // lane half
  const int lr   = lane & 15;
  const int qtl  = wv_ & 3;         // query tile within WG
  const int cs   = wv_ >> 2;        // channel half
  const int b    = blockIdx.z;
  const int i0   = blockIdx.x * 64 + qtl * 16;
  const int cbase = cs * 128;

  const unsigned short* kbase = kbf + (size_t)b * NN * C8;
  const unsigned short* vbase = vbf + (size_t)b * CC * NN;

  // ---- stage chunk j0 into LDS buffer (1152 x 16B units, ~4.5/thread) ----
  auto stage = [&](int j0, int buf) {
    unsigned short* Vd = Vb[buf];
    unsigned short* Kd = Kb[buf];
    #pragma unroll
    for (int r = 0; r < 4; ++r) {
      const int unit = tid + 256 * r;       // 0..1023
      const int c  = unit >> 2;
      const int qd = unit & 3;
      async_cp16(vbase + (size_t)c * NN + j0 + qd * 8, Vd + c * LROW + qd * 8);
    }
    if (tid < 128) {
      const int j  = tid >> 2;
      const int qd = tid & 3;
      async_cp16(kbase + (size_t)(j0 + j) * C8 + qd * 8, Kd + j * LROW + qd * 8);
    }
  };

  // Q^T B-operand (32x16 bf16): lane n = column i, ch 0-7 & 16-23; lane n+16: ch 8-15 & 24-31.
  V16U Qb;
  {
    const unsigned short* qp = qbf + ((size_t)(b * NN + i0 + lr)) * C8 + hl * 8;
    Qb.q4[0] = *(const uint4*)(qp);
    Qb.q4[1] = *(const uint4*)(qp + 16);
  }

  v8f acc[8];
  #pragma unroll
  for (int t = 0; t < 8; ++t)
    acc[t] = (v8f){0.f, 0.f, 0.f, 0.f, 0.f, 0.f, 0.f, 0.f};
  float m = -1e30f, s = 0.f;

  stage(0, 0);
  async_wait0();
  __syncthreads();

  for (int n = 0; n < NCHUNK; ++n) {
    const int buf = n & 1;
    const int j0  = n * 32;
    if (n + 1 < NCHUNK) stage(j0 + 32, buf ^ 1);   // overlap DMA with compute

    // K A-operands from LDS (16-bit A layout: lane<16 ch 0-7 & 16-23, else 8-15 & 24-31)
    const unsigned short* Kl = Kb[buf];
    V16U K0, K1;
    K0.q4[0] = *(const uint4*)(Kl + lr * LROW + hl * 8);
    K0.q4[1] = *(const uint4*)(Kl + lr * LROW + hl * 8 + 16);
    K1.q4[0] = *(const uint4*)(Kl + (16 + lr) * LROW + hl * 8);
    K1.q4[1] = *(const uint4*)(Kl + (16 + lr) * LROW + hl * 8 + 16);

    v8f z = (v8f){0.f, 0.f, 0.f, 0.f, 0.f, 0.f, 0.f, 0.f};
    v8f e0 = __builtin_amdgcn_wmma_f32_16x16x32_bf16(false, K0.v, false, Qb.v,
                                                     (short)0, z, false, false);
    v8f e1 = __builtin_amdgcn_wmma_f32_16x16x32_bf16(false, K1.v, false, Qb.v,
                                                     (short)0, z, false, false);

    // --- online softmax (per lane; merge halves' max via shfl_xor 16) ---
    float cmax = fmaxf(e0[0], e1[0]);
    #pragma unroll
    for (int r = 1; r < 8; ++r) cmax = fmaxf(cmax, fmaxf(e0[r], e1[r]));
    cmax = fmaxf(cmax, __shfl_xor(cmax, 16));
    const float mnew  = fmaxf(m, cmax);
    const float scale = __expf(m - mnew);
    float p0[8], p1[8], psum = 0.f;
    #pragma unroll
    for (int r = 0; r < 8; ++r) { p0[r] = __expf(e0[r] - mnew); psum += p0[r]; }
    #pragma unroll
    for (int r = 0; r < 8; ++r) { p1[r] = __expf(e1[r] - mnew); psum += p1[r]; }
    s = fmaf(s, scale, psum);
    m = mnew;

    // --- repack P into 32x16 bf16 B-operand layout ---
    unsigned pd0[4], pd1[4], od0[4], od1[4];
    #pragma unroll
    for (int r = 0; r < 4; ++r) {
      pd0[r] = pack2bf(p0[2 * r], p0[2 * r + 1]);
      pd1[r] = pack2bf(p1[2 * r], p1[2 * r + 1]);
    }
    #pragma unroll
    for (int r = 0; r < 4; ++r) {
      od0[r] = (unsigned)__shfl_xor((int)pd0[r], 16);
      od1[r] = (unsigned)__shfl_xor((int)pd1[r], 16);
    }
    V16U Pb;
    #pragma unroll
    for (int r = 0; r < 4; ++r) {
      Pb.u[r]     = hl ? od1[r] : pd0[r];
      Pb.u[4 + r] = hl ? pd1[r] : od0[r];
    }

    // --- rescale running output accumulators ---
    #pragma unroll
    for (int t = 0; t < 8; ++t) {
      #pragma unroll
      for (int r = 0; r < 8; ++r) acc[t][r] *= scale;
    }

    // --- O[c,i] += V[c x j] * P^T[j x i] over 8 channel tiles (from LDS) ---
    #pragma unroll
    for (int ct = 0; ct < 8; ++ct) {
      const unsigned short* Vl = Vb[buf] + (cbase + ct * 16 + lr) * LROW + hl * 16;
      V16U Vt;
      Vt.q4[0] = *(const uint4*)(Vl);
      Vt.q4[1] = *(const uint4*)(Vl + 8);
      acc[ct] = __builtin_amdgcn_wmma_f32_16x16x32_bf16(false, Vt.v, false, Pb.v,
                                                        (short)0, acc[ct],
                                                        false, false);
    }

    async_wait0();     // prefetch of chunk n+1 complete
    __syncthreads();   // everyone done reading buf before it is overwritten
  }

  // finalize: merge lane-half sums, apply 1/s, gamma, residual
  const float stot = s + __shfl_xor(s, 16);
  const float sc   = gamma[0] / stot;
  #pragma unroll
  for (int ct = 0; ct < 8; ++ct) {
    #pragma unroll
    for (int r = 0; r < 8; ++r) {
      const int c = cbase + ct * 16 + r + hl * 8;       // C/D layout: M=r(+8)
      const size_t idx = ((size_t)(b * CC + c)) * NN + (i0 + lr);
      out[idx] = fmaf(acc[ct][r], sc, x[idx]);          // gamma*out + x
    }
  }
}

// ---------------------------------------------------------------------------
extern "C" void kernel_launch(void* const* d_in, const int* in_sizes, int n_in,
                              void* d_out, int out_size, void* d_ws, size_t ws_size,
                              hipStream_t stream) {
  const float* x     = (const float*)d_in[0];
  const float* wq    = (const float*)d_in[1];
  const float* bq    = (const float*)d_in[2];
  const float* wk    = (const float*)d_in[3];
  const float* bk    = (const float*)d_in[4];
  const float* wv    = (const float*)d_in[5];
  const float* bv    = (const float*)d_in[6];
  const float* gamma = (const float*)d_in[7];
  float* out = (float*)d_out;

  // workspace: q[B,N,32] bf16 (1MB) | k[B,N,32] bf16 (1MB) | v[B,C,N] bf16 (8MB)
  unsigned short* qbf = (unsigned short*)d_ws;
  unsigned short* kbf = qbf + (size_t)BB * NN * C8;
  unsigned short* vbf = kbf + (size_t)BB * NN * C8;

  qkv_project<<<dim3(NN / 256, 40, BB), 256, 0, stream>>>(
      x, wq, bq, wk, bk, wv, bv, qbf, kbf, vbf);

  flash_attn<<<dim3(NN / 64, 1, BB), 256, 0, stream>>>(
      qbf, kbf, vbf, x, gamma, out);
}
